// IndRNN_1838246003079
// MI455X (gfx1250) — matrix operand order repaired and checked by reading, not compiled, not run on hardware
//
#include <hip/hip_runtime.h>

// ---------------------------------------------------------------------------
// IndRNN on MI455X (gfx1250).
//   Pre-pass : x, W converted f32 -> bf16 once (RNE).
//   Per layer: (1) xh = A_bf16 @ W_bf16^T + b  via v_wmma_f32_16x16x32_bf16,
//                  double-buffered LDS, 128x128 block tile, 32x64 wave tile.
//              (2) scan h_t = relu(xh_t + u*h_{t-1}); emits bf16 for the next
//                  layer's GEMM (f32 to d_out only on the last layer).
// All intermediates live in ws and stay resident in the 192MB L2.
// ---------------------------------------------------------------------------

typedef __bf16 bf16;
typedef __attribute__((ext_vector_type(16))) __bf16   v16bf;
typedef __attribute__((ext_vector_type(8)))  float    v8f;
typedef __attribute__((ext_vector_type(4)))  unsigned u32x4;

#define T_LEN   512
#define BATCH   32
#define D       512                   // D_IN == D_HID
#define M_TOT   (T_LEN * BATCH)       // 16384
#define NBD     (BATCH * D)           // 16384 elements per time step
#define BM      128
#define BN      128
#define BK      32
#define LSTRIDE 40                    // bf16 elems; 80B rows: b128-aligned,
                                      // 20 dwords -> conflict-free frag reads

__device__ __forceinline__ unsigned short bfbits(float f) {
    unsigned u = __builtin_bit_cast(unsigned, f);
    return (unsigned short)((u + 0x7FFFu + ((u >> 16) & 1u)) >> 16);
}
__device__ __forceinline__ bf16 f2bf(float f) {
    return __builtin_bit_cast(bf16, bfbits(f));
}
__device__ __forceinline__ unsigned pack2(float a, float b) {
    return (unsigned)bfbits(a) | ((unsigned)bfbits(b) << 16);
}

// ---------------------------------------------------------------------------
// Bulk f32 -> bf16 conversion: 8 elements per thread, b128 in / b128 out.
// ---------------------------------------------------------------------------
__global__ __launch_bounds__(256)
void conv_bf16(const float* __restrict__ src, bf16* __restrict__ dst, int n8) {
    const int i = blockIdx.x * 256 + threadIdx.x;
    if (i >= n8) return;
    const float4 v0 = ((const float4*)src)[2 * i];
    const float4 v1 = ((const float4*)src)[2 * i + 1];
    u32x4 o;
    o.x = pack2(v0.x, v0.y);
    o.y = pack2(v0.z, v0.w);
    o.z = pack2(v1.x, v1.y);
    o.w = pack2(v1.z, v1.w);
    ((u32x4*)dst)[i] = o;
}

// ---------------------------------------------------------------------------
// GEMM: XH[m,n] = sum_k A[m,k]*B[n,k] + bias[n]   (A,B bf16; XH f32)
// 256 threads = 8 waves (4 in M x 2 in N); wave tile 32x64 = 2x4 WMMA tiles.
// Double-buffered LDS; K-step 32; 16 k-iterations.
// ---------------------------------------------------------------------------
__global__ __launch_bounds__(256)
void indrnn_gemm(const bf16* __restrict__ A, const bf16* __restrict__ B,
                 const float* __restrict__ bias, float* __restrict__ XH) {
    __shared__ bf16 As[2][BM * LSTRIDE];
    __shared__ bf16 Bs[2][BN * LSTRIDE];

    const int tid   = threadIdx.x;
    const int lane  = tid & 31;
    const int w     = tid >> 5;       // wave 0..7
    const int mw    = w & 3;          // M quadrant (x32)
    const int nw    = w >> 2;         // N half     (x64)
    const int row16 = lane & 15;
    const int hi    = lane >> 4;

    const int bm0 = blockIdx.x * BM;
    const int bn0 = blockIdx.y * BN;

    // staging coords: each thread copies 2 A-rows and 2 B-rows, 8 bf16 each
    const int sr = tid >> 2;          // 0..63
    const int sc = (tid & 3) * 8;     // 0,8,16,24

    auto stage = [&](int buf, int k0) {
        const u32x4 a0 = *(const u32x4*)(A + (size_t)(bm0 + sr)      * D + k0 + sc);
        const u32x4 a1 = *(const u32x4*)(A + (size_t)(bm0 + sr + 64) * D + k0 + sc);
        const u32x4 b0 = *(const u32x4*)(B + (size_t)(bn0 + sr)      * D + k0 + sc);
        const u32x4 b1 = *(const u32x4*)(B + (size_t)(bn0 + sr + 64) * D + k0 + sc);
        *(u32x4*)&As[buf][(sr)      * LSTRIDE + sc] = a0;
        *(u32x4*)&As[buf][(sr + 64) * LSTRIDE + sc] = a1;
        *(u32x4*)&Bs[buf][(sr)      * LSTRIDE + sc] = b0;
        *(u32x4*)&Bs[buf][(sr + 64) * LSTRIDE + sc] = b1;
    };

    const v8f vzero = {0.f, 0.f, 0.f, 0.f, 0.f, 0.f, 0.f, 0.f};
    v8f acc[2][4];
    #pragma unroll
    for (int i = 0; i < 2; ++i)
        #pragma unroll
        for (int j = 0; j < 4; ++j) acc[i][j] = vzero;

    union Frag { v16bf v; unsigned u[8]; };

    stage(0, 0);
    __syncthreads();

    const int nk = D / BK;            // 16
    for (int kb = 0; kb < nk; ++kb) {
        const int cur = kb & 1;
        if (kb + 1 < nk) stage(cur ^ 1, (kb + 1) * BK);  // overlap with WMMA

        // ---- fragments from LDS per ISA 7.12.2 layouts ----
        Frag a[2], bf[4];
        #pragma unroll
        for (int i = 0; i < 2; ++i) {
            const int m = mw * 32 + i * 16 + row16;
            #pragma unroll
            for (int p = 0; p < 8; ++p) {
                // A 16x32: pair p -> K = (p>>2)*16 + hi*8 + 2*(p&3)
                const int K = ((p >> 2) * 16) + hi * 8 + 2 * (p & 3);
                a[i].u[p] = *(const unsigned*)&As[cur][m * LSTRIDE + K];
            }
        }
        #pragma unroll
        for (int j = 0; j < 4; ++j) {
            const int n = nw * 64 + j * 16 + row16;
            #pragma unroll
            for (int p = 0; p < 8; ++p) {
                // B 32x16: pair p -> K = (p>>2)*16 + hi*8 + 2*(p&3)
                const int K = ((p >> 2) * 16) + hi * 8 + 2 * (p & 3);
                bf[j].u[p] = *(const unsigned*)&Bs[cur][n * LSTRIDE + K];
            }
        }

        #pragma unroll
        for (int i = 0; i < 2; ++i)
            #pragma unroll
            for (int j = 0; j < 4; ++j)
                acc[i][j] = __builtin_amdgcn_wmma_f32_16x16x32_bf16(
                    false, a[i].v, false, bf[j].v,
                    (short)0, acc[i][j], false, false);

        __syncthreads();
    }

    // ---- epilogue: bias + f32 store (D: VGPR r -> M=r+hi*8, N=row16) ----
    #pragma unroll
    for (int j = 0; j < 4; ++j) {
        const int ng = bn0 + nw * 64 + j * 16 + row16;
        const float bv = bias[ng];
        #pragma unroll
        for (int i = 0; i < 2; ++i) {
            #pragma unroll
            for (int r = 0; r < 8; ++r) {
                const int mg = bm0 + mw * 32 + i * 16 + hi * 8 + r;
                XH[(size_t)mg * D + ng] = acc[i][j][r] + bv;
            }
        }
    }
}

// ---------------------------------------------------------------------------
// Scan: h_t = relu(xh_t + u*h_{t-1}).  Emits bf16 (next layer's GEMM input)
// and/or f32 (final output).  Flags are launch-uniform.
// ---------------------------------------------------------------------------
__global__ __launch_bounds__(256)
void indrnn_scan(const float* __restrict__ xh, const float* __restrict__ u,
                 float* __restrict__ outf, bf16* __restrict__ outb,
                 int write_f32, int write_bf16) {
    const int idx = blockIdx.x * blockDim.x + threadIdx.x;   // 0..16383
    const float uu = u[idx & (D - 1)];
    float h = 0.f;
    #pragma unroll 4
    for (int t = 0; t < T_LEN; ++t) {
        const size_t ofs = (size_t)t * NBD + idx;
        const float v = xh[ofs] + uu * h;
        h = fmaxf(v, 0.f);
        if (write_bf16) outb[ofs] = f2bf(h);
        if (write_f32)  outf[ofs] = h;
    }
}

// ---------------------------------------------------------------------------
extern "C" void kernel_launch(void* const* d_in, const int* in_sizes, int n_in,
                              void* d_out, int out_size, void* d_ws,
                              size_t ws_size, hipStream_t stream) {
    const float* x = (const float*)d_in[0];   // [T, B, D]
    const float* W = (const float*)d_in[1];   // [4, D, D]
    const float* b = (const float*)d_in[2];   // [4, D]
    const float* u = (const float*)d_in[3];   // [4, D]
    float* out = (float*)d_out;               // [T, B, D]

    // ws layout
    char* p = (char*)d_ws;
    float* xh   = (float*)p;            p += (size_t)M_TOT * D * 4;   // 33.5MB
    bf16*  xbf  = (bf16*)p;             p += (size_t)M_TOT * D * 2;   // 16.7MB
    bf16*  actb = (bf16*)p;             p += (size_t)M_TOT * D * 2;   // 16.7MB
    bf16*  Wbf  = (bf16*)p;                                           //  2.0MB

    // one-time conversions (stream-ordered, deterministic every call)
    const int nx8 = (M_TOT * D) / 8;          // 1048576
    const int nw8 = (4 * D * D) / 8;          // 131072
    conv_bf16<<<nx8 / 256, 256, 0, stream>>>(x, xbf, nx8);
    conv_bf16<<<nw8 / 256, 256, 0, stream>>>(W, Wbf, nw8);

    const dim3 gemm_grid(M_TOT / BM, D / BN);     // 128 x 4
    const int  scan_blocks = NBD / 256;           // 64

    const bf16* cur = xbf;
    for (int l = 0; l < 4; ++l) {
        indrnn_gemm<<<gemm_grid, 256, 0, stream>>>(
            cur, Wbf + (size_t)l * D * D, b + (size_t)l * D, xh);
        const int last = (l == 3);
        indrnn_scan<<<scan_blocks, 256, 0, stream>>>(
            xh, u + (size_t)l * D, out, actb, last, !last);
        cur = actb;
    }
}
